// InfoNCELoss_14663018349031
// MI455X (gfx1250) — matrix operand order repaired and checked by reading, not compiled, not run on hardware
//
#include <hip/hip_runtime.h>
#include <hip/hip_bf16.h>
#include <math.h>

typedef __attribute__((ext_vector_type(16))) _Float16 v16h;
typedef __attribute__((ext_vector_type(8)))  _Float16 v8h;
typedef __attribute__((ext_vector_type(4)))  _Float16 v4h;
typedef __attribute__((ext_vector_type(8)))  float    v8f;
typedef __attribute__((ext_vector_type(4)))  float    v4f;

#define DIMS     1024
#define BATCH    4096
#define TEMP_INV 10.0f
#define EPS_N    1e-8f

// ---------------------------------------------------------------------------
// Kernel 1: per-row L2 normalize, fp32 -> f16, one block (256 thr) per row.
// ---------------------------------------------------------------------------
__global__ void __launch_bounds__(256)
normalize_rows_f16(const float* __restrict__ feats, _Float16* __restrict__ zn) {
  __shared__ float red[8];
  const int row = blockIdx.x;
  const int tid = threadIdx.x;
  const float* rp = feats + (size_t)row * DIMS;

  v4f x = *(const v4f*)(rp + tid * 4);              // 256*4 = 1024 elems
  float ss = x.x * x.x + x.y * x.y + x.z * x.z + x.w * x.w;
  #pragma unroll
  for (int off = 16; off >= 1; off >>= 1) ss += __shfl_xor(ss, off, 32);
  if ((tid & 31) == 0) red[tid >> 5] = ss;
  __syncthreads();
  float tot = 0.f;
  #pragma unroll
  for (int i = 0; i < 8; ++i) tot += red[i];
  const float inv = 1.0f / fmaxf(sqrtf(tot), EPS_N);

  v4h o;
  o.x = (_Float16)(x.x * inv);
  o.y = (_Float16)(x.y * inv);
  o.z = (_Float16)(x.z * inv);
  o.w = (_Float16)(x.w * inv);
  *(v4h*)(zn + (size_t)row * DIMS + tid * 4) = o;
}

// ---------------------------------------------------------------------------
// online-softmax fold of a 64-column strip (4 tiles held in x0..x3) into the
// running (max, sumexp) of the row owned by this lane's 16-lane group.
// ---------------------------------------------------------------------------
__device__ __forceinline__ void fold_strip(float x0, float x1, float x2, float x3,
                                           float& m_run, float& s_run) {
  float tm = fmaxf(fmaxf(x0, x1), fmaxf(x2, x3));
  tm = fmaxf(tm, __shfl_xor(tm, 1, 32));
  tm = fmaxf(tm, __shfl_xor(tm, 2, 32));
  tm = fmaxf(tm, __shfl_xor(tm, 4, 32));
  tm = fmaxf(tm, __shfl_xor(tm, 8, 32));
  float te = __expf(x0 - tm) + __expf(x1 - tm) + __expf(x2 - tm) + __expf(x3 - tm);
  te += __shfl_xor(te, 1, 32);
  te += __shfl_xor(te, 2, 32);
  te += __shfl_xor(te, 4, 32);
  te += __shfl_xor(te, 8, 32);
  float nm = fmaxf(m_run, tm);
  s_run = s_run * __expf(m_run - nm) + te * __expf(tm - nm);
  m_run = nm;
}

// ---------------------------------------------------------------------------
// Kernel 2: flash-style logits GEMM (WMMA f16 -> f32) + online softmax stats.
// One workgroup (8 waves) per 32-row z1 block. Each wave computes a 32x64
// output patch per strip (2 row tiles x 4 column tiles): per k-step
// 12 b128 loads feed 8 wmma (1.5 loads/wmma -> XDL-pipe bound).
// 8 strip groups cover all 4096 columns. Logits never hit memory.
//
// WMMA 16x16x32 f16 fragment layouts (ISA 7.12.2, wave32):
//   A (16xK): lane m%16 = row; this code: half 0 holds K 0..15, half 1 K 16..31 (contiguous)
//   B (Kx16): lane n%16 = col (= z2 row for A*B^T); this code: half 0 K 0..15, half 1 K 16..31 (same as A)
//   C/D:      VGPR v, lane l -> (M = v + 8*(l/16), N = l%16)
// ---------------------------------------------------------------------------
#define WMMA_F16(A, B, C) \
  __builtin_amdgcn_wmma_f32_16x16x32_f16(false, (A), false, (B), (short)0, (C), false, false)

#define DIAG_CAP(flag, x, base)                                  \
  if (flag) {                                                    \
    if (lane == v)      sm_diag[(base) + v]     = (x);           \
    if (lane == v + 24) sm_diag[(base) + v + 8] = (x);           \
  }

__global__ void __launch_bounds__(256)
infonce_gemm(const _Float16* __restrict__ z1n,
             const _Float16* __restrict__ z2n,
             float* __restrict__ rowloss) {
  __shared__ float sm_m[8][32];
  __shared__ float sm_s[8][32];
  __shared__ float sm_diag[32];

  const int wave = threadIdx.x >> 5;
  const int lane = threadIdx.x & 31;
  const int lm   = lane & 15;
  const int lh   = lane >> 4;
  const int r0   = blockIdx.x << 5;          // this block's 32 z1 rows

  const _Float16* arow0 = z1n + (size_t)(r0 + lm) * DIMS;
  const _Float16* arow1 = z1n + (size_t)(r0 + 16 + lm) * DIMS;
  const int akb = lh * 16;                   // A per-lane K sub-offset
  const int bkb = lh * 16;                   // B per-lane K sub-offset

  float m0[8], s0[8], m1[8], s1[8];
  #pragma unroll
  for (int v = 0; v < 8; ++v) {
    m0[v] = -1e30f; s0[v] = 0.f;
    m1[v] = -1e30f; s1[v] = 0.f;
  }

  const int d0t = blockIdx.x * 2;            // diag col-tile for row-tile 0
  const int d1t = d0t + 1;                   // diag col-tile for row-tile 1

  // 256 column tiles / (8 waves * 4 tiles) = 8 strip groups per wave
  for (int jg = 0; jg < 8; ++jg) {
    const int jb = wave * 4 + jg * 32;       // first of 4 consecutive tiles
    const _Float16* br0 = z2n + (size_t)((jb + 0) * 16 + lm) * DIMS + bkb;
    const _Float16* br1 = z2n + (size_t)((jb + 1) * 16 + lm) * DIMS + bkb;
    const _Float16* br2 = z2n + (size_t)((jb + 2) * 16 + lm) * DIMS + bkb;
    const _Float16* br3 = z2n + (size_t)((jb + 3) * 16 + lm) * DIMS + bkb;

    v8f c00 = {}, c01 = {}, c02 = {}, c03 = {};
    v8f c10 = {}, c11 = {}, c12 = {}, c13 = {};
    #pragma unroll 1
    for (int k0 = 0; k0 < DIMS; k0 += 32) {
      v8h a0lo = *(const v8h*)(arow0 + k0 + akb);
      v8h a0hi = *(const v8h*)(arow0 + k0 + akb + 8);
      v8h a1lo = *(const v8h*)(arow1 + k0 + akb);
      v8h a1hi = *(const v8h*)(arow1 + k0 + akb + 8);
      v16h a0 = __builtin_shufflevector(a0lo, a0hi,
                  0, 1, 2, 3, 4, 5, 6, 7, 8, 9, 10, 11, 12, 13, 14, 15);
      v16h a1 = __builtin_shufflevector(a1lo, a1hi,
                  0, 1, 2, 3, 4, 5, 6, 7, 8, 9, 10, 11, 12, 13, 14, 15);
      v16h b0 = *(const v16h*)(br0 + k0);
      v16h b1 = *(const v16h*)(br1 + k0);
      v16h b2 = *(const v16h*)(br2 + k0);
      v16h b3 = *(const v16h*)(br3 + k0);
      c00 = WMMA_F16(a0, b0, c00);
      c01 = WMMA_F16(a0, b1, c01);
      c02 = WMMA_F16(a0, b2, c02);
      c03 = WMMA_F16(a0, b3, c03);
      c10 = WMMA_F16(a1, b0, c10);
      c11 = WMMA_F16(a1, b1, c11);
      c12 = WMMA_F16(a1, b2, c12);
      c13 = WMMA_F16(a1, b3, c13);
    }

    // uniform per wave: which (if any) tiles hold the two diagonal blocks
    const bool h00 = (jb + 0 == d0t), h01 = (jb + 1 == d0t),
               h02 = (jb + 2 == d0t), h03 = (jb + 3 == d0t);
    const bool h10 = (jb + 0 == d1t), h11 = (jb + 1 == d1t),
               h12 = (jb + 2 == d1t), h13 = (jb + 3 == d1t);

    #pragma unroll
    for (int v = 0; v < 8; ++v) {
      // ---- row tile 0 (rows r0 .. r0+15) ----
      {
        float x0 = c00[v] * TEMP_INV;
        float x1 = c01[v] * TEMP_INV;
        float x2 = c02[v] * TEMP_INV;
        float x3 = c03[v] * TEMP_INV;
        DIAG_CAP(h00, x0, 0) DIAG_CAP(h01, x1, 0)
        DIAG_CAP(h02, x2, 0) DIAG_CAP(h03, x3, 0)
        fold_strip(x0, x1, x2, x3, m0[v], s0[v]);
      }
      // ---- row tile 1 (rows r0+16 .. r0+31) ----
      {
        float x0 = c10[v] * TEMP_INV;
        float x1 = c11[v] * TEMP_INV;
        float x2 = c12[v] * TEMP_INV;
        float x3 = c13[v] * TEMP_INV;
        DIAG_CAP(h10, x0, 16) DIAG_CAP(h11, x1, 16)
        DIAG_CAP(h12, x2, 16) DIAG_CAP(h13, x3, 16)
        fold_strip(x0, x1, x2, x3, m1[v], s1[v]);
      }
    }
  }

  // publish per-wave stats (lane 0 carries rows 0..7, lane 16 rows 8..15)
  if (lm == 0) {
    #pragma unroll
    for (int v = 0; v < 8; ++v) {
      sm_m[wave][lh * 8 + v]      = m0[v];
      sm_s[wave][lh * 8 + v]      = s0[v];
      sm_m[wave][16 + lh * 8 + v] = m1[v];
      sm_s[wave][16 + lh * 8 + v] = s1[v];
    }
  }
  __syncthreads();

  // combine 8 waves, emit per-row loss: -(l_ii - max - log(sumexp))
  if (threadIdx.x < 32) {
    const int r = threadIdx.x;
    float mt = -1e30f;
    #pragma unroll
    for (int w = 0; w < 8; ++w) mt = fmaxf(mt, sm_m[w][r]);
    float st = 0.f;
    #pragma unroll
    for (int w = 0; w < 8; ++w) st += sm_s[w][r] * __expf(sm_m[w][r] - mt);
    rowloss[r0 + r] = -(sm_diag[r] - mt - __logf(st));
  }
}

// ---------------------------------------------------------------------------
// Kernel 3: deterministic mean over 4096 per-row losses.
// ---------------------------------------------------------------------------
__global__ void __launch_bounds__(256)
final_reduce(const float* __restrict__ rowloss, float* __restrict__ out) {
  __shared__ float red[256];
  float s = 0.f;
  for (int i = threadIdx.x; i < BATCH; i += 256) s += rowloss[i];
  red[threadIdx.x] = s;
  __syncthreads();
  for (int off = 128; off >= 1; off >>= 1) {
    if (threadIdx.x < off) red[threadIdx.x] += red[threadIdx.x + off];
    __syncthreads();
  }
  if (threadIdx.x == 0) out[0] = red[0] / (float)BATCH;
}

// ---------------------------------------------------------------------------
extern "C" void kernel_launch(void* const* d_in, const int* in_sizes, int n_in,
                              void* d_out, int out_size, void* d_ws, size_t ws_size,
                              hipStream_t stream) {
  (void)in_sizes; (void)n_in; (void)out_size; (void)ws_size;
  const float* feats = (const float*)d_in[0];

  _Float16* zn      = (_Float16*)d_ws;                         // 8192x1024 f16
  _Float16* z1n     = zn;                                      // rows 0..4095
  _Float16* z2n     = zn + (size_t)BATCH * DIMS;               // rows 4096..8191
  float*    rowloss = (float*)((char*)d_ws +
                        (size_t)2 * BATCH * DIMS * sizeof(_Float16));
  float*    out     = (float*)d_out;

  normalize_rows_f16<<<2 * BATCH, 256, 0, stream>>>(feats, zn);
  infonce_gemm<<<BATCH / 32, 256, 0, stream>>>(z1n, z2n, rowloss);
  final_reduce<<<1, 256, 0, stream>>>(rowloss, out);
}
